// ModelOp_35699768164968
// MI455X (gfx1250) — matrix operand order repaired and checked by reading, not compiled, not run on hardware
//
#include <hip/hip_runtime.h>

#define HID 256
#define QQ  64

typedef __attribute__((ext_vector_type(16))) __bf16 v16bf;
typedef __attribute__((ext_vector_type(8)))  float  v8f;

union Frag16 { v16bf v; unsigned u[8]; };

__device__ __forceinline__ unsigned short f2bf(float f) {
  unsigned u = __float_as_uint(f);
  u = u + 0x7FFFu + ((u >> 16) & 1u);      // round-to-nearest-even
  return (unsigned short)(u >> 16);
}
__device__ __forceinline__ float bf2f(unsigned short s) {
  return __uint_as_float(((unsigned)s) << 16);
}
__device__ __forceinline__ unsigned lds_addr(const void* p) {
  return (unsigned)(unsigned long long)p;   // addrspace(3) offset
}

// ---------------------------------------------------------------------------
// Tiled WMMA GEMM: C[z] = act(A @ Bt[z]^T + bias)
//   A : [M,K]  fp32 or bf16(ushort)
//   Bt: [NN,K] bf16(ushort), pre-transposed weights, z-batched
//   C : [M,NN] fp32 and/or bf16
// Block tile 128x64, 8 waves (4x2), wave tile 32x32 via 2x2 WMMA, K-step 32.
// Full K-steps: unguarded vectorized loads; bf16 paths use
// GLOBAL_LOAD_ASYNC_TO_LDS_B128 (ASYNCcnt) straight into LDS.
// ---------------------------------------------------------------------------
template<bool ABF16, int ACT, bool OUTF, bool OUTB>
__global__ __launch_bounds__(256)
void gemm_kernel(const void* __restrict__ Aptr, int lda,
                 const unsigned short* __restrict__ Btw, int ldBt, long long strideB,
                 const float* __restrict__ bias,
                 float* __restrict__ Cf, unsigned short* __restrict__ Cb,
                 int ldC, long long strideC,
                 int M, int K, int NN)
{
  constexpr int BM = 128, BN = 64, BK = 32, AST = 40, BST = 40;
  __shared__ unsigned short As[BM * AST];        // [row][k], 80B row pitch
  __shared__ unsigned short Bs[BN * BST];        // [n][k],   80B row pitch

  const int z = blockIdx.z;
  const unsigned short* Bt = Btw + (size_t)z * (size_t)strideB;
  const int tid  = threadIdx.x;
  const int wave = tid >> 5, lane = tid & 31;
  const int wm = wave >> 1, wn = wave & 1;       // 4 x 2 wave grid
  const int row0 = blockIdx.y * BM, col0 = blockIdx.x * BN;
  const int hlf = lane >> 4, ln = lane & 15;

  v8f acc[2][2] = {};

  for (int kb = 0; kb < K; kb += BK) {
    __syncthreads();
    const bool full = (kb + BK <= K);            // uniform branch
    if (full) {
      // ---------------- fast path ----------------
      if constexpr (ABF16) {
        // A tile: 128x32 bf16 = 8KB, 2 async b128 per thread
        #pragma unroll
        for (int i = 0; i < 2; ++i) {
          int slot = tid + i * 256;              // 512 slots x 8 ushorts
          int r  = slot >> 2;
          int ko = (slot & 3) * 8;
          int gr = row0 + r; if (gr >= M) gr = M - 1;   // clamp; masked at store
          const unsigned short* gp = (const unsigned short*)Aptr + (size_t)gr * lda + kb + ko;
          unsigned lp = lds_addr(&As[r * AST + ko]);
          asm volatile("global_load_async_to_lds_b128 %0, %1, off"
                       :: "v"(lp), "v"(gp) : "memory");
        }
      } else {
        // A tile fp32 -> bf16: 4 float4 loads + b64 LDS stores per thread
        #pragma unroll
        for (int i = 0; i < 4; ++i) {
          int slot = tid + i * 256;              // 1024 slots x 4 floats
          int r  = slot >> 3;
          int kq = (slot & 7) * 4;
          int gr = row0 + r; if (gr >= M) gr = M - 1;
          float4 v = *(const float4*)((const float*)Aptr + (size_t)gr * lda + kb + kq);
          uint2 p;
          p.x = (unsigned)f2bf(v.x) | ((unsigned)f2bf(v.y) << 16);
          p.y = (unsigned)f2bf(v.z) | ((unsigned)f2bf(v.w) << 16);
          *(uint2*)&As[r * AST + kq] = p;
        }
        if (kb + BK < K) {
          int gr = row0 + (tid >> 1);
          if (gr < M) __builtin_prefetch((const float*)Aptr + (size_t)gr * lda + kb + BK, 0, 1);
        }
      }
      // B tile: 64x32 bf16 = 4KB, 1 async b128 per thread
      {
        int n  = tid >> 2;
        int ko = (tid & 3) * 8;
        int gn = col0 + n; if (gn >= NN) gn = NN - 1;
        const unsigned short* gp = Bt + (size_t)gn * ldBt + kb + ko;
        unsigned lp = lds_addr(&Bs[n * BST + ko]);
        asm volatile("global_load_async_to_lds_b128 %0, %1, off"
                     :: "v"(lp), "v"(gp) : "memory");
      }
      asm volatile("s_wait_asynccnt 0x0" ::: "memory");
    } else {
      // ---------------- guarded tail (K=5/3 projections only) ----------------
      #pragma unroll
      for (int i = 0; i < 16; ++i) {
        int e = tid + i * 256;
        int r = e >> 5, kk = e & 31;
        int gr = row0 + r; if (gr >= M) gr = M - 1;
        int gk = kb + kk;
        unsigned short bv = 0;
        if (gk < K) {
          if constexpr (ABF16) bv = ((const unsigned short*)Aptr)[(size_t)gr * lda + gk];
          else                 bv = f2bf(((const float*)Aptr)[(size_t)gr * lda + gk]);
        }
        As[r * AST + kk] = bv;
      }
      #pragma unroll
      for (int i = 0; i < 8; ++i) {
        int e = tid + i * 256;
        int n = e >> 5, kk = e & 31;
        int gn = col0 + n; if (gn >= NN) gn = NN - 1;
        int gk = kb + kk;
        Bs[n * BST + kk] = (gk < K) ? Bt[(size_t)gn * ldBt + gk] : (unsigned short)0;
      }
    }
    __syncthreads();

    // ---- fragments in contiguous k order (k = 16*half + s), 2x2 WMMA
    Frag16 bfr[2];
    #pragma unroll
    for (int tn = 0; tn < 2; ++tn) {
      int n = wn * 32 + tn * 16 + ln;
      #pragma unroll
      for (int i = 0; i < 8; ++i) {
        int bk = hlf * 16 + 2 * i;               // B: lanes0-15 K=0..15, lanes16-31 K=16..31
        bfr[tn].u[i] = *(const unsigned*)&Bs[n * BST + bk];
      }
    }
    #pragma unroll
    for (int tm = 0; tm < 2; ++tm) {
      Frag16 afr;
      int m = wm * 32 + tm * 16 + ln;
      #pragma unroll
      for (int i = 0; i < 8; ++i) {
        // A: lanes0-15 K=0..15, lanes16-31 K=16..31
        int bk = hlf * 16 + 2 * i;
        afr.u[i] = *(const unsigned*)&As[m * AST + bk];
      }
      #pragma unroll
      for (int tn = 0; tn < 2; ++tn) {
        acc[tm][tn] = __builtin_amdgcn_wmma_f32_16x16x32_bf16(
            false, afr.v, false, bfr[tn].v, (short)0, acc[tm][tn], false, false);
      }
    }
  }

  // ---- epilogue: bias + activation + store (C layout: VGPR j -> M = j + 8*half)
  #pragma unroll
  for (int tm = 0; tm < 2; ++tm) {
    #pragma unroll
    for (int tn = 0; tn < 2; ++tn) {
      int gn = col0 + wn * 32 + tn * 16 + ln;
      float bval = (bias != nullptr && gn < NN) ? bias[gn] : 0.f;
      #pragma unroll
      for (int j = 0; j < 8; ++j) {
        int gm = row0 + wm * 32 + tm * 16 + hlf * 8 + j;
        if (gm < M && gn < NN) {
          float v = acc[tm][tn][j] + bval;
          if constexpr (ACT == 1) v = v > 0.f ? v : 0.f;         // relu
          if constexpr (ACT == 2) v = v > 0.f ? v : 0.01f * v;   // leaky relu
          size_t off = (size_t)z * (size_t)strideC + (size_t)gm * ldC + gn;
          if constexpr (OUTF) Cf[off] = v;
          if constexpr (OUTB) Cb[off] = f2bf(v);
        }
      }
    }
  }
}

// ---------------------------------------------------------------------------
// weight convert + transpose: src [K,NN] fp32 -> dst [NN,K] bf16
__global__ void cvt_t_kernel(const float* __restrict__ s, unsigned short* __restrict__ d,
                             int K, int NN) {
  int i = blockIdx.x * blockDim.x + threadIdx.x;
  if (i < K * NN) {
    int k = i / NN, n = i - k * NN;
    d[(size_t)n * K + k] = f2bf(s[i]);
  }
}

__global__ void zero_kernel(float* __restrict__ p, long long n) {
  long long i  = (long long)blockIdx.x * blockDim.x + threadIdx.x;
  long long st = (long long)gridDim.x * blockDim.x;
  for (; i < n; i += st) p[i] = 0.f;
}

__global__ void count_kernel(const int* __restrict__ dst, const int* __restrict__ et,
                             float* __restrict__ cnt, int E, int N) {
  int e = blockIdx.x * blockDim.x + threadIdx.x;
  if (e < E) atomicAdd(&cnt[(size_t)et[e] * N + dst[e]], 1.0f);
}

// one block handles 16 edges; 256 lanes move one full 1KB feature row per edge
__global__ __launch_bounds__(256)
void scatter_kernel(const int* __restrict__ src, const int* __restrict__ dst,
                    const int* __restrict__ et, const float* __restrict__ h,
                    float* __restrict__ sums, int E, int N) {
  int e0 = blockIdx.x * 16;
  int hh = threadIdx.x;
  for (int i = 0; i < 16; ++i) {
    int e = e0 + i;
    if (e >= E) return;
    int s = src[e], d = dst[e], r = et[e];
    float v = h[((size_t)r * N + s) * HID + hh];
    atomicAdd(&sums[((size_t)r * N + d) * HID + hh], v);
  }
}

__global__ void combine_kernel(const float* __restrict__ sums, const float* __restrict__ cnt,
                               const float* __restrict__ hroot,
                               unsigned short* __restrict__ xb, int N) {
  long long idx = (long long)blockIdx.x * blockDim.x + threadIdx.x;
  long long tot = (long long)N * HID;
  if (idx >= tot) return;
  int n = (int)(idx >> 8);
  float c0 = cnt[n], c1 = cnt[N + n];
  float v = sums[idx] / fmaxf(c0, 1.f) + sums[tot + idx] / fmaxf(c1, 1.f) + hroot[idx];
  xb[idx] = f2bf(v);
}

// 256 -> 2 head: one wave per node, wave32 shuffle reduction
__global__ __launch_bounds__(256)
void head_kernel(const unsigned short* __restrict__ x, const float* __restrict__ w,
                 const float* __restrict__ b, float* __restrict__ out, int N) {
  int lane = threadIdx.x & 31, wv = threadIdx.x >> 5;
  int n = blockIdx.x * 8 + wv;
  if (n >= N) return;
  float a0 = 0.f, a1 = 0.f;
  for (int k = lane; k < HID; k += 32) {
    float xv = bf2f(x[(size_t)n * HID + k]);
    a0 += xv * w[k * 2 + 0];
    a1 += xv * w[k * 2 + 1];
  }
  #pragma unroll
  for (int off = 16; off >= 1; off >>= 1) {
    a0 += __shfl_down(a0, off, 32);
    a1 += __shfl_down(a1, off, 32);
  }
  if (lane == 0) {
    out[(size_t)n * 2 + 0] = a0 + b[0];
    out[(size_t)n * 2 + 1] = a1 + b[1];
  }
}

// ---------------------------------------------------------------------------
extern "C" void kernel_launch(void* const* d_in, const int* in_sizes, int n_in,
                              void* d_out, int out_size, void* d_ws, size_t ws_size,
                              hipStream_t stream) {
  const float* des   = (const float*)d_in[0];
  const float* tweet = (const float*)d_in[1];
  const float* nump  = (const float*)d_in[2];
  const float* catp  = (const float*)d_in[3];
  const int*   eidx  = (const int*)d_in[4];
  const int*   etyp  = (const int*)d_in[5];
  const float* w_des = (const float*)d_in[6];  const float* b_des = (const float*)d_in[7];
  const float* w_tw  = (const float*)d_in[8];  const float* b_tw  = (const float*)d_in[9];
  const float* w_num = (const float*)d_in[10]; const float* b_num = (const float*)d_in[11];
  const float* w_cat = (const float*)d_in[12]; const float* b_cat = (const float*)d_in[13];
  const float* w_in  = (const float*)d_in[14]; const float* b_in  = (const float*)d_in[15];
  const float* weight1 = (const float*)d_in[16];
  const float* root1   = (const float*)d_in[17]; const float* bias1 = (const float*)d_in[18];
  const float* weight2 = (const float*)d_in[19];
  const float* root2   = (const float*)d_in[20]; const float* bias2 = (const float*)d_in[21];
  const float* w_out   = (const float*)d_in[22]; const float* b_out = (const float*)d_in[23];

  const int DESd = 768, TWd = 768, NPd = 5, CPd = 3, R = 2;
  const int N = in_sizes[0] / DESd;
  const int E = in_sizes[5];
  const int* srcp = eidx;
  const int* dstp = eidx + E;

  // bf16 transposed weight region layout (element offsets), each block [NN][K]
  const size_t O_WDES = 0;
  const size_t O_WTW  = O_WDES + (size_t)DESd * QQ;
  const size_t O_WNUM = O_WTW  + (size_t)TWd * QQ;
  const size_t O_WCAT = O_WNUM + (size_t)NPd * QQ;
  const size_t O_WIN  = O_WCAT + (size_t)CPd * QQ;
  const size_t O_L1   = O_WIN  + (size_t)HID * HID;          // [w1_r0^T, w1_r1^T, root1^T]
  const size_t O_L2   = O_L1   + (size_t)3 * HID * HID;      // [w2_r0^T, w2_r1^T, root2^T]
  const size_t WB_ELEMS = O_L2 + (size_t)3 * HID * HID;

  char* ws = (char*)d_ws;
  auto alignup = [](size_t x) { return (x + 255) & ~(size_t)255; };
  size_t o = 0;
  unsigned short* wb   = (unsigned short*)(ws + o); o = alignup(o + WB_ELEMS * 2);
  unsigned short* actA = (unsigned short*)(ws + o); o = alignup(o + (size_t)N * HID * 2);
  unsigned short* actB = (unsigned short*)(ws + o); o = alignup(o + (size_t)N * HID * 2);
  float* h3   = (float*)(ws + o); o = alignup(o + (size_t)3 * N * HID * 4);  // rel0,rel1,root
  float* sums = (float*)(ws + o); o = alignup(o + (size_t)R * N * HID * 4);
  float* cnt  = (float*)(ws + o); o = alignup(o + (size_t)R * N * 4);        // contiguous after sums
  (void)ws_size; (void)n_in; (void)out_size;

  dim3 blk(256);
  auto cvt = [&](const float* s, size_t off, int K, int NN) {
    size_t n = (size_t)K * NN;
    cvt_t_kernel<<<dim3((unsigned)((n + 255) / 256)), blk, 0, stream>>>(s, wb + off, K, NN);
  };
  // 1) weights -> bf16, transposed to [NN][K]
  cvt(w_des, O_WDES, DESd, QQ);
  cvt(w_tw,  O_WTW,  TWd,  QQ);
  cvt(w_num, O_WNUM, NPd,  QQ);
  cvt(w_cat, O_WCAT, CPd,  QQ);
  cvt(w_in,  O_WIN,  HID,  HID);
  cvt(weight1,                     O_L1,                         HID, HID);   // r0
  cvt(weight1 + (size_t)HID * HID, O_L1 + (size_t)HID * HID,     HID, HID);   // r1
  cvt(root1,                       O_L1 + (size_t)2 * HID * HID, HID, HID);
  cvt(weight2,                     O_L2,                         HID, HID);
  cvt(weight2 + (size_t)HID * HID, O_L2 + (size_t)HID * HID,     HID, HID);
  cvt(root2,                       O_L2 + (size_t)2 * HID * HID, HID, HID);

  unsigned gy = (unsigned)((N + 127) / 128);

  // 2) four projections -> actA (= xcat), lrelu, bf16 columns
  gemm_kernel<false,2,false,true><<<dim3(1,gy,1), blk, 0, stream>>>(
      des,   DESd, wb + O_WDES, DESd, 0, b_des, (float*)nullptr, actA + 0*QQ, HID, 0, N, DESd, QQ);
  gemm_kernel<false,2,false,true><<<dim3(1,gy,1), blk, 0, stream>>>(
      tweet, TWd,  wb + O_WTW,  TWd,  0, b_tw,  (float*)nullptr, actA + 1*QQ, HID, 0, N, TWd, QQ);
  gemm_kernel<false,2,false,true><<<dim3(1,gy,1), blk, 0, stream>>>(
      nump,  NPd,  wb + O_WNUM, NPd,  0, b_num, (float*)nullptr, actA + 2*QQ, HID, 0, N, NPd, QQ);
  gemm_kernel<false,2,false,true><<<dim3(1,gy,1), blk, 0, stream>>>(
      catp,  CPd,  wb + O_WCAT, CPd,  0, b_cat, (float*)nullptr, actA + 3*QQ, HID, 0, N, CPd, QQ);

  // 3) x1 = lrelu(xcat @ w_in + b_in) -> actB
  gemm_kernel<true,2,false,true><<<dim3(4,gy,1), blk, 0, stream>>>(
      actA, HID, wb + O_WIN, HID, 0, b_in, (float*)nullptr, actB, HID, 0, N, HID, HID);

  long long nzero = (long long)R * N * HID + (long long)R * N;  // sums + cnt (contiguous)

  // 4) RGCN layer 1: actB -> actA
  zero_kernel<<<dim3(2048), blk, 0, stream>>>(sums, nzero);
  gemm_kernel<true,1,true,false><<<dim3(4,gy,3), blk, 0, stream>>>(
      actB, HID, wb + O_L1, HID, (long long)HID * HID, bias1,
      h3, (unsigned short*)nullptr, HID, (long long)N * HID, N, HID, HID);
  count_kernel<<<dim3((unsigned)((E + 255) / 256)), blk, 0, stream>>>(dstp, etyp, cnt, E, N);
  scatter_kernel<<<dim3((unsigned)((E + 15) / 16)), blk, 0, stream>>>(srcp, dstp, etyp, h3, sums, E, N);
  combine_kernel<<<dim3((unsigned)(((size_t)N * HID + 255) / 256)), blk, 0, stream>>>(
      sums, cnt, h3 + (size_t)2 * N * HID, actA, N);

  // 5) RGCN layer 2: actA -> actB
  zero_kernel<<<dim3(2048), blk, 0, stream>>>(sums, nzero);
  gemm_kernel<true,1,true,false><<<dim3(4,gy,3), blk, 0, stream>>>(
      actA, HID, wb + O_L2, HID, (long long)HID * HID, bias2,
      h3, (unsigned short*)nullptr, HID, (long long)N * HID, N, HID, HID);
  count_kernel<<<dim3((unsigned)((E + 255) / 256)), blk, 0, stream>>>(dstp, etyp, cnt, E, N);
  scatter_kernel<<<dim3((unsigned)((E + 15) / 16)), blk, 0, stream>>>(srcp, dstp, etyp, h3, sums, E, N);
  combine_kernel<<<dim3((unsigned)(((size_t)N * HID + 255) / 256)), blk, 0, stream>>>(
      sums, cnt, h3 + (size_t)2 * N * HID, actB, N);

  // 6) logits = x @ w_out + b_out
  head_kernel<<<dim3((unsigned)((N + 7) / 8)), blk, 0, stream>>>(actB, w_out, b_out, (float*)d_out, N);
}
